// CGNN_86045374808283
// MI455X (gfx1250) — matrix operand order repaired and checked by reading, not compiled, not run on hardware
//
#include <hip/hip_runtime.h>
#include <hip/hip_bf16.h>

typedef __attribute__((ext_vector_type(16))) __bf16 v16bf;
typedef __attribute__((ext_vector_type(8)))  __bf16 v8bf;
typedef __attribute__((ext_vector_type(8)))  float  v8f;

#define IN_DIM 256
#define HID 64
#define HEADS 4
#define OUT_DIM 16

// ---------- helpers ----------
__device__ __forceinline__ unsigned f2mono(float f) {
    unsigned u = __float_as_uint(f);
    return (u & 0x80000000u) ? ~u : (u | 0x80000000u);
}
__device__ __forceinline__ float mono2f(unsigned u) {
    return (u & 0x80000000u) ? __uint_as_float(u & 0x7fffffffu) : __uint_as_float(~u);
}

// ---------- K0: weight prep (transpose to B-friendly [Ncol][K] bf16, pad, normalize protos) ----------
__global__ void cgnn_prep(const float* __restrict__ W_in, const float* __restrict__ W_l,
                          const float* __restrict__ W_r, const float* __restrict__ b_l,
                          const float* __restrict__ b_r, const float* __restrict__ protos,
                          __bf16* __restrict__ Wt_in, __bf16* __restrict__ Wt_lr,
                          float* __restrict__ b_lr, float* __restrict__ pn)
{
    int idx = blockIdx.x * blockDim.x + threadIdx.x;
    if (idx < 64 * 256) {                       // Wt_in[n][k] = W_in[k][n]   (W_in is [256,64])
        int n = idx >> 8, k = idx & 255;
        Wt_in[n * 256 + k] = (__bf16)W_in[k * 64 + n];
    } else if (idx < 64 * 256 + 512 * 128) {    // Wt_lr[n][k], k padded 66->128, n: 0-255=W_l, 256-511=W_r
        int i = idx - 64 * 256;
        int n = i >> 7, k = i & 127;
        float v = 0.f;
        if (k < 66) v = (n < 256) ? W_l[k * 256 + n] : W_r[k * 256 + (n - 256)];
        Wt_lr[n * 128 + k] = (__bf16)v;
    } else if (idx < 64 * 256 + 512 * 128 + 512) {
        int i = idx - 64 * 256 - 512 * 128;
        b_lr[i] = (i < 256) ? b_l[i] : b_r[i - 256];
    } else if (idx < 64 * 256 + 512 * 128 + 512 + 2) {
        int p = idx - 64 * 256 - 512 * 128 - 512;
        float ss = 0.f;
        for (int c = 0; c < HID; c++) { float v = protos[p * HID + c]; ss += v * v; }
        float inv = 1.f / (sqrtf(ss) + 1e-12f);
        for (int c = 0; c < HID; c++) pn[p * HID + c] = protos[p * HID + c] * inv;
    }
}

// ---------- K1: h = relu(x @ W_in + b_in), bf16 WMMA w/ f32 accumulate ----------
__global__ void cgnn_gemm1(const float* __restrict__ x, const __bf16* __restrict__ Wt,
                           const float* __restrict__ b_in, float* __restrict__ h, int nrows)
{
    int wave = blockIdx.x * (blockDim.x >> 5) + (threadIdx.x >> 5);
    int lane = threadIdx.x & 31;
    int r0 = wave * 16;
    if (r0 >= nrows) return;                    // wave-uniform: EXEC stays all-ones
    int half = lane >> 4, m = lane & 15;
    int rr = r0 + m; if (rr >= nrows) rr = nrows - 1;
    const float* xrow = x + (size_t)rr * IN_DIM;

    v8f acc[4] = {};
    for (int kt = 0; kt < IN_DIM; kt += 32) {
        // A: 16 bf16 per lane = run [kt+half*16, +16)
        float af[16];
        const float4* p0 = reinterpret_cast<const float4*>(xrow + kt + half * 16);
        const float4* p1 = reinterpret_cast<const float4*>(xrow + kt + half * 16 + 8);
        float4 f0 = p0[0], f1 = p0[1], f2 = p1[0], f3 = p1[1];
        af[0]=f0.x; af[1]=f0.y; af[2]=f0.z; af[3]=f0.w;
        af[4]=f1.x; af[5]=f1.y; af[6]=f1.z; af[7]=f1.w;
        af[8]=f2.x; af[9]=f2.y; af[10]=f2.z; af[11]=f2.w;
        af[12]=f3.x; af[13]=f3.y; af[14]=f3.z; af[15]=f3.w;
        v16bf A;
        #pragma unroll
        for (int i = 0; i < 16; i++) A[i] = (__bf16)af[i];
        #pragma unroll
        for (int n = 0; n < 4; n++) {
            v16bf B = *reinterpret_cast<const v16bf*>(Wt + (size_t)(n * 16 + m) * IN_DIM + kt + half * 16);
            acc[n] = __builtin_amdgcn_wmma_f32_16x16x32_bf16(false, A, false, B, (short)0, acc[n], false, false);
        }
    }
    #pragma unroll
    for (int n = 0; n < 4; n++) {
        int col = n * 16 + m;
        float bias = b_in[col];
        #pragma unroll
        for (int j = 0; j < 8; j++) {
            int row = r0 + j + 8 * half;
            if (row < nrows) {
                float v = acc[n][j] + bias;
                h[(size_t)row * HID + col] = v > 0.f ? v : 0.f;
            }
        }
    }
}

// ---------- K2: cosine sim to prototypes, build bf16 h_aug[N][128] (66 used, rest 0) ----------
__global__ void cgnn_proto(const float* __restrict__ h, const float* __restrict__ pn,
                           __bf16* __restrict__ haug, int n)
{
    int node = blockIdx.x * (blockDim.x >> 5) + (threadIdx.x >> 5);
    int lane = threadIdx.x & 31;
    if (node >= n) return;
    float v0 = h[(size_t)node * HID + 2 * lane];
    float v1 = h[(size_t)node * HID + 2 * lane + 1];
    float ss = v0 * v0 + v1 * v1;
    float d0 = v0 * pn[2 * lane] + v1 * pn[2 * lane + 1];
    float d1 = v0 * pn[HID + 2 * lane] + v1 * pn[HID + 2 * lane + 1];
    #pragma unroll
    for (int msk = 16; msk >= 1; msk >>= 1) {
        ss += __shfl_xor(ss, msk);
        d0 += __shfl_xor(d0, msk);
        d1 += __shfl_xor(d1, msk);
    }
    float inv = 1.f / (sqrtf(ss) + 1e-12f);
    __bf16* row = haug + (size_t)node * 128;
    row[2 * lane] = (__bf16)v0;
    row[2 * lane + 1] = (__bf16)v1;
    if (lane == 0) {
        row[64] = (__bf16)(d0 * inv);
        row[65] = (__bf16)(d1 * inv);
    } else {
        row[64 + 2 * lane] = (__bf16)0.f;
        row[64 + 2 * lane + 1] = (__bf16)0.f;
    }
}

// ---------- K3: xlr = h_aug @ [W_l|W_r] + b  -> bf16 [N][512] (xl cols 0-255, xr 256-511) ----------
__global__ void cgnn_gemm2(const __bf16* __restrict__ haug, const __bf16* __restrict__ Wt,
                           const float* __restrict__ b_lr, __bf16* __restrict__ xlr, int nrows)
{
    int wave = blockIdx.x * (blockDim.x >> 5) + (threadIdx.x >> 5);
    int lane = threadIdx.x & 31;
    int rb = wave >> 2, tg = wave & 3;          // 4 tile-groups of 8 N-tiles (128 cols) each
    int r0 = rb * 16;
    if (r0 >= nrows) return;
    int half = lane >> 4, m = lane & 15;
    int rr = r0 + m; if (rr >= nrows) rr = nrows - 1;
    const __bf16* arow = haug + (size_t)rr * 128;

    v8f acc[8] = {};
    for (int kt = 0; kt < 128; kt += 32) {
        v8bf a0 = *reinterpret_cast<const v8bf*>(arow + kt + half * 16);
        v8bf a1 = *reinterpret_cast<const v8bf*>(arow + kt + half * 16 + 8);
        v16bf A;
        #pragma unroll
        for (int i = 0; i < 8; i++) { A[i] = a0[i]; A[i + 8] = a1[i]; }
        #pragma unroll
        for (int t = 0; t < 8; t++) {
            int nc = (tg * 8 + t) * 16 + m;
            v16bf B = *reinterpret_cast<const v16bf*>(Wt + (size_t)nc * 128 + kt + half * 16);
            acc[t] = __builtin_amdgcn_wmma_f32_16x16x32_bf16(false, A, false, B, (short)0, acc[t], false, false);
        }
    }
    #pragma unroll
    for (int t = 0; t < 8; t++) {
        int col = (tg * 8 + t) * 16 + m;
        float bias = b_lr[col];
        #pragma unroll
        for (int j = 0; j < 8; j++) {
            int row = r0 + j + 8 * half;
            if (row < nrows) xlr[(size_t)row * 512 + col] = (__bf16)(acc[t][j] + bias);
        }
    }
}

// ---------- K4: edge logits + segment max (wave per edge; 8 lanes per head) ----------
__global__ void cgnn_logits(const __bf16* __restrict__ xlr, const int* __restrict__ ei,
                            const float* __restrict__ att, float* __restrict__ logits,
                            unsigned* __restrict__ segmax, int E, int n)
{
    int e = blockIdx.x * (blockDim.x >> 5) + (threadIdx.x >> 5);
    int lane = threadIdx.x & 31;
    int ET = E + n;
    if (e >= ET) return;
    int src, dst;
    if (e < E) { src = ei[e]; dst = ei[E + e]; } else { src = dst = e - E; }
    int h = lane >> 3, i = lane & 7;
    int off = h * HID + i * 8;
    v8bf bl = *reinterpret_cast<const v8bf*>(xlr + (size_t)src * 512 + off);
    v8bf br = *reinterpret_cast<const v8bf*>(xlr + (size_t)dst * 512 + 256 + off);
    const float* pa = att + off;
    float s = 0.f;
    #pragma unroll
    for (int j = 0; j < 8; j++) {
        float v = (float)bl[j] + (float)br[j];
        v = v > 0.f ? v : 0.2f * v;             // leaky_relu, slope 0.2
        s += v * pa[j];
    }
    s += __shfl_xor(s, 1); s += __shfl_xor(s, 2); s += __shfl_xor(s, 4);
    if (i == 0) {
        logits[(size_t)e * HEADS + h] = s;
        atomicMax(segmax + (size_t)dst * HEADS + h, f2mono(s));
    }
}

// ---------- K5: alpha = exp(logit - max), accumulate denominators ----------
__global__ void cgnn_expsum(const int* __restrict__ ei, float* __restrict__ logits,
                            const unsigned* __restrict__ segmax, float* __restrict__ denom,
                            int E, int n)
{
    int idx = blockIdx.x * blockDim.x + threadIdx.x;
    int ET = E + n;
    if (idx >= ET * HEADS) return;
    int e = idx >> 2, h = idx & 3;
    int dst = (e < E) ? ei[E + e] : (e - E);
    float mx = mono2f(segmax[(size_t)dst * HEADS + h]);
    float a = __expf(logits[idx] - mx);
    logits[idx] = a;                            // reuse buffer for exp values
    unsafeAtomicAdd(denom + (size_t)dst * HEADS + h, a);
}

// ---------- K6: weighted scatter-add of messages (wave per edge) ----------
__global__ void cgnn_scatter(const __bf16* __restrict__ xlr, const int* __restrict__ ei,
                             const float* __restrict__ expv, const float* __restrict__ denom,
                             float* __restrict__ agg, int E, int n)
{
    int e = blockIdx.x * (blockDim.x >> 5) + (threadIdx.x >> 5);
    int lane = threadIdx.x & 31;
    int ET = E + n;
    if (e >= ET) return;
    int src, dst;
    if (e < E) { src = ei[e]; dst = ei[E + e]; } else { src = dst = e - E; }
    int h = lane >> 3, i = lane & 7;
    float w = expv[(size_t)e * HEADS + h] / (denom[(size_t)dst * HEADS + h] + 1e-16f);
    int off = h * HID + i * 8;
    v8bf bl = *reinterpret_cast<const v8bf*>(xlr + (size_t)src * 512 + off);
    float* pagg = agg + (size_t)dst * (HEADS * HID) + off;
    #pragma unroll
    for (int j = 0; j < 8; j++) unsafeAtomicAdd(pagg + j, (float)bl[j] * w);
}

// ---------- K7: head mean + bias + relu + @W_cls + b_cls (wave per node) ----------
__global__ void cgnn_final(const float* __restrict__ agg, const float* __restrict__ gat_bias,
                           const float* __restrict__ Wc, const float* __restrict__ b_cls,
                           float* __restrict__ out, int n)
{
    int node = blockIdx.x * (blockDim.x >> 5) + (threadIdx.x >> 5);
    int lane = threadIdx.x & 31;
    if (node >= n) return;
    int c0 = 2 * lane, c1 = c0 + 1;
    float s0 = 0.f, s1 = 0.f;
    const float* arow = agg + (size_t)node * (HEADS * HID);
    #pragma unroll
    for (int h = 0; h < HEADS; h++) { s0 += arow[h * HID + c0]; s1 += arow[h * HID + c1]; }
    float t0 = s0 * 0.25f + gat_bias[c0]; t0 = t0 > 0.f ? t0 : 0.f;
    float t1 = s1 * 0.25f + gat_bias[c1]; t1 = t1 > 0.f ? t1 : 0.f;
    int o = lane & 15;
    float accv = 0.f;
    #pragma unroll
    for (int c = 0; c < HID; c++) {
        float tc = __shfl((c & 1) ? t1 : t0, c >> 1);
        accv += tc * Wc[c * OUT_DIM + o];
    }
    if (lane < 16) out[(size_t)node * OUT_DIM + lane] = accv + b_cls[lane];
}

// ---------- launch ----------
extern "C" void kernel_launch(void* const* d_in, const int* in_sizes, int n_in,
                              void* d_out, int out_size, void* d_ws, size_t ws_size,
                              hipStream_t stream)
{
    const float* x      = (const float*)d_in[0];
    const int*   ei     = (const int*)d_in[1];
    const float* W_in   = (const float*)d_in[2];
    const float* b_in   = (const float*)d_in[3];
    const float* protos = (const float*)d_in[4];
    const float* W_l    = (const float*)d_in[5];
    const float* b_l    = (const float*)d_in[6];
    const float* W_r    = (const float*)d_in[7];
    const float* b_r    = (const float*)d_in[8];
    const float* att    = (const float*)d_in[9];
    const float* gbias  = (const float*)d_in[10];
    const float* W_cls  = (const float*)d_in[11];
    const float* b_cls  = (const float*)d_in[12];
    float* out = (float*)d_out;

    const int N  = in_sizes[0] / IN_DIM;   // 50000
    const int E  = in_sizes[1] / 2;        // 800000
    const int ET = E + N;

    size_t off = 0;
    auto take = [&](size_t bytes) { void* p = (char*)d_ws + off; off += (bytes + 255) & ~(size_t)255; return p; };
    float*    h_buf  = (float*)   take((size_t)N * HID * 4);
    __bf16*   haug   = (__bf16*)  take((size_t)N * 128 * 2);
    __bf16*   xlr    = (__bf16*)  take((size_t)N * 512 * 2);
    float*    logits = (float*)   take((size_t)ET * HEADS * 4);
    unsigned* segmax = (unsigned*)take((size_t)N * HEADS * 4);
    float*    denom  = (float*)   take((size_t)N * HEADS * 4);
    float*    agg    = (float*)   take((size_t)N * HEADS * HID * 4);
    __bf16*   Wt_in  = (__bf16*)  take(64 * 256 * 2);
    __bf16*   Wt_lr  = (__bf16*)  take(512 * 128 * 2);
    float*    b_lr   = (float*)   take(512 * 4);
    float*    pn     = (float*)   take(2 * HID * 4);

    // prep weights (tiny)
    {
        int total = 64 * 256 + 512 * 128 + 512 + 2;
        cgnn_prep<<<(total + 255) / 256, 256, 0, stream>>>(W_in, W_l, W_r, b_l, b_r, protos,
                                                           Wt_in, Wt_lr, b_lr, pn);
    }
    // zero accumulators (capturable async memsets)
    hipMemsetAsync(segmax, 0, (size_t)N * HEADS * 4, stream);   // 0 == mono(-NaN), below all reals
    hipMemsetAsync(denom, 0, (size_t)N * HEADS * 4, stream);
    hipMemsetAsync(agg, 0, (size_t)N * HEADS * HID * 4, stream);

    const int RB = (N + 15) / 16;  // 16-row blocks
    // K1: GEMM1 (1 wave per 16x64 tile)
    cgnn_gemm1<<<(RB + 7) / 8, 256, 0, stream>>>(x, Wt_in, b_in, h_buf, N);
    // K2: prototypes + h_aug
    cgnn_proto<<<(N + 7) / 8, 256, 0, stream>>>(h_buf, pn, haug, N);
    // K3: GEMM2 (4 waves per row-block, each 16x128 of output)
    cgnn_gemm2<<<(RB * 4 + 7) / 8, 256, 0, stream>>>(haug, Wt_lr, b_lr, xlr, N);
    // K4: logits + segment max
    cgnn_logits<<<(ET + 7) / 8, 256, 0, stream>>>(xlr, ei, att, logits, segmax, E, N);
    // K5: exp + denominators
    cgnn_expsum<<<((size_t)ET * HEADS + 255) / 256, 256, 0, stream>>>(ei, logits, segmax, denom, E, N);
    // K6: weighted scatter aggregation
    cgnn_scatter<<<(ET + 7) / 8, 256, 0, stream>>>(xlr, ei, logits, denom, agg, E, N);
    // K7: finalize
    cgnn_final<<<(N + 7) / 8, 256, 0, stream>>>(agg, gbias, W_cls, b_cls, out, N);
}
